// LiquidLM_14199161880585
// MI455X (gfx1250) — matrix operand order repaired and checked by reading, not compiled, not run on hardware
//
#include <hip/hip_runtime.h>
#include <hip/hip_bf16.h>

// ---- problem dims ----
constexpr int Bq  = 2;
constexpr int Tq  = 512;
constexpr int Dq  = 1024;
constexpr int Vq  = 32000;
constexpr int Nq  = 128;
constexpr int NHq = 8;
constexpr int HDq = 128;
constexpr int BTq = Bq * Tq;

// ---- WMMA GEMM tiling ----
constexpr int BM = 128;
constexpr int BN = 128;
constexpr int BK = 32;
constexpr int LDS_STRIDE = 40; // ushort stride (32 + 8 pad), 80B rows stay 16B aligned

enum { GF_BIAS = 1, GF_ACCUM = 2, GF_RELU = 4 };

typedef __bf16 v16bf __attribute__((ext_vector_type(16)));
typedef float  v8f   __attribute__((ext_vector_type(8)));
typedef unsigned int u32x8 __attribute__((ext_vector_type(8)));

__device__ __forceinline__ unsigned short f2bf(float f) {
  union { float f; unsigned u; } c; c.f = f;
  unsigned r = c.u + 0x7FFFu + ((c.u >> 16) & 1u); // round-to-nearest-even
  return (unsigned short)(r >> 16);
}

__device__ __forceinline__ v16bf frag_from_lds(const unsigned short* p0,
                                               const unsigned short* p1) {
  uint4 a = *reinterpret_cast<const uint4*>(p0);
  uint4 b = *reinterpret_cast<const uint4*>(p1);
  u32x8 t;
  t[0] = a.x; t[1] = a.y; t[2] = a.z; t[3] = a.w;
  t[4] = b.x; t[5] = b.y; t[6] = b.z; t[7] = b.w;
  return __builtin_bit_cast(v16bf, t);
}

__device__ __forceinline__ void cvt_store16(unsigned short* d, const float4* r) {
  d[0]  = f2bf(r[0].x); d[1]  = f2bf(r[0].y); d[2]  = f2bf(r[0].z); d[3]  = f2bf(r[0].w);
  d[4]  = f2bf(r[1].x); d[5]  = f2bf(r[1].y); d[6]  = f2bf(r[1].z); d[7]  = f2bf(r[1].w);
  d[8]  = f2bf(r[2].x); d[9]  = f2bf(r[2].y); d[10] = f2bf(r[2].z); d[11] = f2bf(r[2].w);
  d[12] = f2bf(r[3].x); d[13] = f2bf(r[3].y); d[14] = f2bf(r[3].z); d[15] = f2bf(r[3].w);
}

// C[M,N] = A[M,K] * W[N,K]^T  (+ bias[N]) (+= C) (relu), flags compile-time.
// Requires M%128==0, N%128==0, K%32==0. 256 threads = 8 waves; wave owns a
// 32x64 C sub-tile = 2x4 v_wmma_f32_16x16x32_bf16 per K-step. Next K-tile is
// prefetched into registers while the current one is staged to LDS + consumed
// by the WMMAs, hiding global latency behind matrix ops.
template <int FLAGS>
__global__ __launch_bounds__(256)
void gemm_bf16_wmma(const float* __restrict__ A, const float* __restrict__ W,
                    const float* __restrict__ bias, float* __restrict__ C,
                    int M, int N, int K) {
  __shared__ unsigned short sA[BM * LDS_STRIDE];
  __shared__ unsigned short sB[BN * LDS_STRIDE];

  const int tid  = threadIdx.x;
  const int lane = tid & 31;
  const int wave = tid >> 5;
  const int wm   = (wave & 3) * 32;   // wave row offset in block tile (4 row groups)
  const int wn   = (wave >> 2) * 64;  // wave col offset in block tile (2 col groups)
  const long m0  = (long)blockIdx.y * BM;
  const long n0  = (long)blockIdx.x * BN;

  const int lr = lane & 15;   // row (A) / col (B,C) within a 16-wide tile
  const int hi = lane >> 4;   // half-wave select

  v8f acc[2][4];
  if (FLAGS & GF_ACCUM) {
    for (int tm = 0; tm < 2; ++tm)
      for (int tn = 0; tn < 4; ++tn)
        for (int r = 0; r < 8; ++r) {
          long row = m0 + wm + tm * 16 + hi * 8 + r;
          long col = n0 + wn + tn * 16 + lr;
          acc[tm][tn][r] = C[row * (long)N + col];
        }
  } else {
    for (int tm = 0; tm < 2; ++tm)
      for (int tn = 0; tn < 4; ++tn)
        for (int r = 0; r < 8; ++r) acc[tm][tn][r] = 0.f;
  }

  // cooperative tile loaders: A and B tiles are both 128x32 f32 -> 16 f32/thread
  const int ldRow = tid >> 1;
  const int ldCol = (tid & 1) * 16;
  const float* aPtr = A + (m0 + ldRow) * (long)K + ldCol;
  const float* bPtr = W + (n0 + ldRow) * (long)K + ldCol;
  unsigned short* da = &sA[ldRow * LDS_STRIDE + ldCol];
  unsigned short* db = &sB[ldRow * LDS_STRIDE + ldCol];

  float4 ca[4], cb[4], na[4], nb[4];
  for (int i = 0; i < 4; ++i) {
    ca[i] = *reinterpret_cast<const float4*>(aPtr + i * 4);
    cb[i] = *reinterpret_cast<const float4*>(bPtr + i * 4);
  }

  const int ac = hi * 8;    // A frag K-half offsets {0,8}, second chunk +16
  const int bc = hi * 8;    // B frag K-half offsets {0,8}, second chunk +16

  for (int k0 = 0; k0 < K; k0 += BK) {
    const bool more = (k0 + BK) < K;
    if (more) {  // register prefetch of next K-tile, overlaps LDS + WMMA below
      for (int i = 0; i < 4; ++i) {
        na[i] = *reinterpret_cast<const float4*>(aPtr + k0 + BK + i * 4);
        nb[i] = *reinterpret_cast<const float4*>(bPtr + k0 + BK + i * 4);
      }
    }

    cvt_store16(da, ca);
    cvt_store16(db, cb);
    __syncthreads();

    // A frag (16x32 MxK): lanes0-15 K {0..7,16..23}; lanes16-31 K {8..15,24..31}
    // B frag (32x16 KxN): lanes0-15 K {0..7,16..23}; lanes16-31 K {8..15,24..31}
    v16bf af[2], bfr[4];
    for (int tm = 0; tm < 2; ++tm) {
      const unsigned short* base = &sA[(wm + tm * 16 + lr) * LDS_STRIDE];
      af[tm] = frag_from_lds(base + ac, base + ac + 16);
    }
    for (int tn = 0; tn < 4; ++tn) {
      const unsigned short* base = &sB[(wn + tn * 16 + lr) * LDS_STRIDE];
      bfr[tn] = frag_from_lds(base + bc, base + bc + 16);
    }
    for (int tm = 0; tm < 2; ++tm)
      for (int tn = 0; tn < 4; ++tn)
        acc[tm][tn] = __builtin_amdgcn_wmma_f32_16x16x32_bf16(
            false, af[tm], false, bfr[tn], (short)0, acc[tm][tn], false, false);
    __syncthreads();

    if (more) {
      for (int i = 0; i < 4; ++i) { ca[i] = na[i]; cb[i] = nb[i]; }
    }
  }

  for (int tn = 0; tn < 4; ++tn) {
    long col = n0 + wn + tn * 16 + lr;
    float bv = (FLAGS & GF_BIAS) ? bias[col] : 0.f;
    for (int tm = 0; tm < 2; ++tm)
      for (int r = 0; r < 8; ++r) {
        long row = m0 + wm + tm * 16 + hi * 8 + r;
        float v = acc[tm][tn][r] + bv;
        if (FLAGS & GF_RELU) v = fmaxf(v, 0.f);
        C[row * (long)N + col] = v;
      }
  }
}

// ---- embedding gather ----
__global__ __launch_bounds__(256)
void embed_k(const int* __restrict__ toks, const float* __restrict__ emb,
             float* __restrict__ x) {
  const int bt = blockIdx.x;
  const int tok = toks[bt];
  for (int d = threadIdx.x; d < Dq; d += 256)
    x[(size_t)bt * Dq + d] = emb[(size_t)tok * Dq + d];
}

// ---- elementwise add ----
__global__ __launch_bounds__(256)
void add_k(const float* __restrict__ a, const float* __restrict__ b,
           float* __restrict__ c, int n) {
  int i = blockIdx.x * 256 + threadIdx.x;
  if (i < n) c[i] = a[i] + b[i];
}

// ---- S4D diagonal SSM scan: h_t = h_{t-1}*decay + Bu_t ----
__global__ __launch_bounds__(256)
void ssm_scan_k(const float* __restrict__ Bu, const float* __restrict__ log_lambda,
                const float* __restrict__ log_dt, float* __restrict__ hs) {
  const int tid = threadIdx.x;          // 256 = B*N
  const int b = tid >> 7, n = tid & 127;
  const float decay = __expf(-__expf(log_lambda[n]) * __expf(log_dt[0]));
  float h = 0.f;
  const float* bu = Bu + (size_t)b * Tq * Nq + n;
  float* o = hs + (size_t)b * Tq * Nq + n;
  for (int t = 0; t < Tq; ++t) { h = h * decay + bu[t * Nq]; o[t * Nq] = h; }
}

// ---- LayerNorm over D, in-place ----
__global__ __launch_bounds__(256)
void layernorm_k(float* __restrict__ x, const float* __restrict__ g,
                 const float* __restrict__ b) {
  __shared__ float red[256];
  const int tid = threadIdx.x;
  float* row = x + (size_t)blockIdx.x * Dq;
  float v[4];
  float s = 0.f;
  for (int i = 0; i < 4; ++i) { v[i] = row[tid + i * 256]; s += v[i]; }
  red[tid] = s; __syncthreads();
  for (int st = 128; st > 0; st >>= 1) { if (tid < st) red[tid] += red[tid + st]; __syncthreads(); }
  const float mu = red[0] / Dq; __syncthreads();
  float q = 0.f;
  for (int i = 0; i < 4; ++i) { float d = v[i] - mu; q += d * d; }
  red[tid] = q; __syncthreads();
  for (int st = 128; st > 0; st >>= 1) { if (tid < st) red[tid] += red[tid + st]; __syncthreads(); }
  const float inv = rsqrtf(red[0] / Dq + 1e-5f);
  for (int i = 0; i < 4; ++i) {
    int d = tid + i * 256;
    row[d] = (v[i] - mu) * inv * g[d] + b[d];
  }
}

__device__ __forceinline__ float phi_elu1(float z) {
  return z > 0.f ? z + 1.f : __expf(z);  // elu(z)+1
}

// ---- linear attention: per (b,head) reduce k_sum and kv over T ----
__global__ __launch_bounds__(128)
void attn_kv_k(const float* __restrict__ k_lin, const float* __restrict__ v_lin,
               float* __restrict__ ksum, float* __restrict__ kvs) {
  const int bh = blockIdx.x;            // B*NH
  const int b = bh / NHq, h = bh % NHq;
  const int hd = threadIdx.x;           // HD = 128
  float ks = 0.f, kv = 0.f;
  for (int t = 0; t < Tq; ++t) {
    size_t idx = ((size_t)(b * Tq + t)) * Dq + h * HDq + hd;
    float kp = phi_elu1(k_lin[idx]);
    ks += kp;
    kv += kp * v_lin[idx];
  }
  ksum[(size_t)bh * HDq + hd] = ks;
  kvs[(size_t)bh * HDq + hd] = kv;
}

// ---- attention output + sigmoid gate blend: one wave32 per head ----
__global__ __launch_bounds__(256)
void attn_out_k(const float* __restrict__ q_lin, const float* __restrict__ gate_lin,
                const float* __restrict__ xo, const float* __restrict__ ksum,
                const float* __restrict__ kvs, float* __restrict__ hout) {
  const int bt = blockIdx.x;
  const int b = bt / Tq;
  const int lane = threadIdx.x & 31;
  const int h = threadIdx.x >> 5;       // 8 waves = 8 heads
  const float* qrow = q_lin + (size_t)bt * Dq + h * HDq;
  const float* ks = ksum + (size_t)(b * NHq + h) * HDq;
  const float* kv = kvs + (size_t)(b * NHq + h) * HDq;
  float s = 0.f, z = 0.f;
  for (int j = 0; j < 4; ++j) {
    int hd = lane + j * 32;
    float qv = phi_elu1(qrow[hd]);
    s += qv * kv[hd];
    z += qv * ks[hd];
  }
  for (int off = 16; off >= 1; off >>= 1) {
    s += __shfl_xor(s, off, 32);
    z += __shfl_xor(z, off, 32);
  }
  const float outv = s / (z + 1e-6f);
  for (int j = 0; j < 4; ++j) {
    int d = h * HDq + lane + j * 32;
    size_t idx = (size_t)bt * Dq + d;
    float g = 1.f / (1.f + __expf(-gate_lin[idx]));
    hout[idx] = g * outv + (1.f - g) * xo[idx];
  }
}

// ---- softmax over V per token ----
__global__ __launch_bounds__(256)
void softmax_v_k(const float* __restrict__ logits, float* __restrict__ probs) {
  __shared__ float red[256];
  const int tid = threadIdx.x;
  const size_t base = (size_t)blockIdx.x * Vq;
  float mx = -1e30f;
  for (int i = tid; i < Vq; i += 256) mx = fmaxf(mx, logits[base + i]);
  red[tid] = mx; __syncthreads();
  for (int st = 128; st > 0; st >>= 1) { if (tid < st) red[tid] = fmaxf(red[tid], red[tid + st]); __syncthreads(); }
  const float m = red[0]; __syncthreads();
  float sm = 0.f;
  for (int i = tid; i < Vq; i += 256) sm += __expf(logits[base + i] - m);
  red[tid] = sm; __syncthreads();
  for (int st = 128; st > 0; st >>= 1) { if (tid < st) red[tid] += red[tid + st]; __syncthreads(); }
  const float inv = 1.f / red[0];
  for (int i = tid; i < Vq; i += 256) probs[base + i] = __expf(logits[base + i] - m) * inv;
}

// ---- row dot with vector w + scalar bias; optional sigmoid ----
__global__ __launch_bounds__(256)
void rowdot_k(const float* __restrict__ X, const float* __restrict__ w,
              const float* __restrict__ bscal, float* __restrict__ out, int act) {
  __shared__ float red[256];
  const int tid = threadIdx.x;
  const float* row = X + (size_t)blockIdx.x * Dq;
  float s = 0.f;
  for (int i = tid; i < Dq; i += 256) s += row[i] * w[i];
  red[tid] = s; __syncthreads();
  for (int st = 128; st > 0; st >>= 1) { if (tid < st) red[tid] += red[tid + st]; __syncthreads(); }
  if (tid == 0) {
    float r = red[0] + bscal[0];
    if (act) r = 1.f / (1.f + __expf(-r));
    out[blockIdx.x] = r;
  }
}

// ---- softmax over T per batch (T=512, 2 per thread) ----
__global__ __launch_bounds__(256)
void softmax_t_k(const float* __restrict__ s, float* __restrict__ w) {
  __shared__ float red[256];
  const int tid = threadIdx.x;
  const float* sr = s + (size_t)blockIdx.x * Tq;
  float a = sr[tid], b = sr[tid + 256];
  red[tid] = fmaxf(a, b); __syncthreads();
  for (int st = 128; st > 0; st >>= 1) { if (tid < st) red[tid] = fmaxf(red[tid], red[tid + st]); __syncthreads(); }
  const float m = red[0]; __syncthreads();
  float e0 = __expf(a - m), e1 = __expf(b - m);
  red[tid] = e0 + e1; __syncthreads();
  for (int st = 128; st > 0; st >>= 1) { if (tid < st) red[tid] += red[tid + st]; __syncthreads(); }
  const float inv = 1.f / red[0];
  w[(size_t)blockIdx.x * Tq + tid] = e0 * inv;
  w[(size_t)blockIdx.x * Tq + tid + 256] = e1 * inv;
}

// ---- weighted pooled sum over T: out[b,d] = sum_t w[b,t]*p[b,t,d] ----
__global__ __launch_bounds__(256)
void wsum_k(const float* __restrict__ wts, const float* __restrict__ p,
            float* __restrict__ out) {
  const int b = blockIdx.y;
  const int d = blockIdx.x * 256 + threadIdx.x;
  float acc = 0.f;
  for (int t = 0; t < Tq; ++t)
    acc += wts[b * Tq + t] * p[((size_t)(b * Tq + t)) * Dq + d];
  out[(size_t)b * Dq + d] = acc;
}

extern "C" void kernel_launch(void* const* d_in, const int* in_sizes, int n_in,
                              void* d_out, int out_size, void* d_ws, size_t ws_size,
                              hipStream_t stream) {
  const int*   tokens    = (const int*)  d_in[0];
  const float* emb       = (const float*)d_in[1];
  const float* in_w      = (const float*)d_in[2];
  const float* log_lam   = (const float*)d_in[3];
  const float* B_ssm     = (const float*)d_in[4];
  const float* C_ssm     = (const float*)d_in[5];
  const float* log_dt    = (const float*)d_in[6];
  const float* ssm_out_w = (const float*)d_in[7];
  const float* ln_g      = (const float*)d_in[8];
  const float* ln_b      = (const float*)d_in[9];
  const float* q_w       = (const float*)d_in[10];
  const float* k_w       = (const float*)d_in[11];
  const float* v_w       = (const float*)d_in[12];
  const float* gate_w    = (const float*)d_in[13];
  const float* gate_b    = (const float*)d_in[14];
  const float* head_w    = (const float*)d_in[15];
  const float* head_b    = (const float*)d_in[16];
  const float* sp_w      = (const float*)d_in[17];
  const float* sp_b      = (const float*)d_in[18];
  const float* ap_w      = (const float*)d_in[19];
  const float* ap_b      = (const float*)d_in[20];
  const float* op_w      = (const float*)d_in[21];
  const float* op_b      = (const float*)d_in[22];
  const float* ss_w      = (const float*)d_in[23];
  const float* ss_b      = (const float*)d_in[24];
  const float* as_w      = (const float*)d_in[25];
  const float* as_b      = (const float*)d_in[26];
  const float* os_w      = (const float*)d_in[27];
  const float* os_b      = (const float*)d_in[28];
  const float* fc1_wh    = (const float*)d_in[29];
  const float* fc1_wv    = (const float*)d_in[30];
  const float* fc1_b     = (const float*)d_in[31];
  const float* fc2_w     = (const float*)d_in[32];
  const float* fc2_b     = (const float*)d_in[33];

  // output layout: logits [B,T,V] | conf [B,T] | subj [B,D] | act [B,D] | obj [B,D]
  float* logits = (float*)d_out;
  float* conf   = logits + (size_t)BTq * Vq;
  float* subj   = conf + BTq;
  float* actp   = subj + (size_t)Bq * Dq;
  float* objp   = actp + (size_t)Bq * Dq;

  // workspace bump allocation
  float* w = (float*)d_ws;
  float* x      = w; w += (size_t)BTq * Dq;
  float* xi     = w; w += (size_t)BTq * Dq;
  float* Bu     = w; w += (size_t)BTq * Nq;
  float* hs     = w; w += (size_t)BTq * Nq;
  float* ybuf   = w; w += (size_t)BTq * Dq;
  float* xo     = w; w += (size_t)BTq * Dq;
  float* qb     = w; w += (size_t)BTq * Dq;
  float* kb     = w; w += (size_t)BTq * Dq;
  float* vb     = w; w += (size_t)BTq * Dq;
  float* gb     = w; w += (size_t)BTq * Dq;
  float* hbuf   = w; w += (size_t)BTq * Dq;
  float* ksum   = w; w += (size_t)Bq * NHq * HDq;
  float* kvs    = w; w += (size_t)Bq * NHq * HDq;
  float* probs  = w; w += (size_t)BTq * Vq;
  float* hid    = w; w += (size_t)BTq * Dq;
  float* scores = w; w += BTq;
  float* wts    = w; w += BTq;
  float* pbuf   = w; w += (size_t)BTq * Dq;

  const dim3 blk(256);
  auto launch_gemm = [&](const float* A, const float* Wt, const float* bias,
                         float* Cp, int M, int N, int K, int flags) {
    dim3 g(N / BN, M / BM);
    if (flags == 0)
      gemm_bf16_wmma<0><<<g, blk, 0, stream>>>(A, Wt, bias, Cp, M, N, K);
    else if (flags == GF_BIAS)
      gemm_bf16_wmma<GF_BIAS><<<g, blk, 0, stream>>>(A, Wt, bias, Cp, M, N, K);
    else
      gemm_bf16_wmma<GF_ACCUM | GF_RELU><<<g, blk, 0, stream>>>(A, Wt, bias, Cp, M, N, K);
  };

  // ---- Mamba block ----
  embed_k<<<BTq, blk, 0, stream>>>(tokens, emb, x);
  launch_gemm(x, in_w, nullptr, xi, BTq, Dq, Dq, 0);
  launch_gemm(xi, B_ssm, nullptr, Bu, BTq, Nq, Dq, 0);
  ssm_scan_k<<<1, blk, 0, stream>>>(Bu, log_lam, log_dt, hs);
  launch_gemm(hs, C_ssm, nullptr, ybuf, BTq, Dq, Nq, 0);
  add_k<<<(BTq * Dq) / 256, blk, 0, stream>>>(xi, ybuf, ybuf, BTq * Dq);
  launch_gemm(ybuf, ssm_out_w, nullptr, xo, BTq, Dq, Dq, 0);
  layernorm_k<<<BTq, blk, 0, stream>>>(xo, ln_g, ln_b);

  // ---- liquid self-attention ----
  launch_gemm(xo, q_w, nullptr, qb, BTq, Dq, Dq, 0);
  launch_gemm(xo, k_w, nullptr, kb, BTq, Dq, Dq, 0);
  launch_gemm(xo, v_w, nullptr, vb, BTq, Dq, Dq, 0);
  launch_gemm(xo, gate_w, gate_b, gb, BTq, Dq, Dq, GF_BIAS);
  attn_kv_k<<<Bq * NHq, dim3(128), 0, stream>>>(kb, vb, ksum, kvs);
  attn_out_k<<<BTq, blk, 0, stream>>>(qb, gb, xo, ksum, kvs, hbuf);

  // ---- LM head + softmax ----
  launch_gemm(hbuf, head_w, head_b, logits, BTq, Vq, Dq, GF_BIAS);
  softmax_v_k<<<BTq, blk, 0, stream>>>(logits, probs);

  // ---- confidence net: hid = relu(h@fc1_wh^T + probs@fc1_wv^T + fc1_b) ----
  launch_gemm(hbuf, fc1_wh, fc1_b, hid, BTq, Dq, Dq, GF_BIAS);
  launch_gemm(probs, fc1_wv, nullptr, hid, BTq, Dq, Vq, GF_ACCUM | GF_RELU);
  rowdot_k<<<BTq, blk, 0, stream>>>(hid, fc2_w, fc2_b, conf, 1);

  // ---- concept extractor (subject / action / object) ----
  const float* sw[3] = { ss_w, as_w, os_w };
  const float* sb[3] = { ss_b, as_b, os_b };
  const float* pw[3] = { sp_w, ap_w, op_w };
  const float* pb[3] = { sp_b, ap_b, op_b };
  float* outc[3] = { subj, actp, objp };
  for (int c = 0; c < 3; ++c) {
    rowdot_k<<<BTq, blk, 0, stream>>>(hbuf, sw[c], sb[c], scores, 0);
    softmax_t_k<<<Bq, blk, 0, stream>>>(scores, wts);
    launch_gemm(hbuf, pw[c], pb[c], pbuf, BTq, Dq, Dq, GF_BIAS);
    wsum_k<<<dim3(Dq / 256, Bq), blk, 0, stream>>>(wts, pbuf, outc[c]);
  }
}
